// BasicResidualBlock_21655225106980
// MI455X (gfx1250) — matrix operand order repaired and checked by reading, not compiled, not run on hardware
//
#include <hip/hip_runtime.h>

typedef __bf16 bf16_t;
typedef __attribute__((ext_vector_type(16))) __bf16 v16bf;
typedef __attribute__((ext_vector_type(8)))  float  v8f;
typedef __attribute__((address_space(3))) unsigned char lds_uchar;

#define EPS_BN 1e-5f

// Async global -> LDS 16-byte copy (CDNA5 ASYNCcnt path, bypasses VGPRs)
__device__ __forceinline__ void async_copy16(void* lds_dst, const void* gsrc) {
    unsigned lds = (unsigned)(size_t)(lds_uchar*)lds_dst;
    unsigned long long ga = (unsigned long long)gsrc;
    asm volatile("global_load_async_to_lds_b128 %0, %1, off"
                 :: "v"(lds), "v"(ga) : "memory");
}
__device__ __forceinline__ void wait_async0() {
    asm volatile("s_wait_asynccnt 0x0" ::: "memory");
}

// ---------------------------------------------------------------------------
// Weight prep: w[co][ci][3][3] f32 -> Whi/Wlo[tap][co][ci] bf16 (two-term split)
// ---------------------------------------------------------------------------
__global__ void prep_weights(const float* __restrict__ w,
                             bf16_t* __restrict__ whi, bf16_t* __restrict__ wlo) {
    int idx = blockIdx.x * 256 + threadIdx.x;          // (tap*128+co)*128+ci
    if (idx >= 9 * 128 * 128) return;
    int ci  = idx & 127;
    int co  = (idx >> 7) & 127;
    int tap = idx >> 14;
    float val = w[(co * 128 + ci) * 9 + tap];
    bf16_t hi = (bf16_t)val;
    bf16_t lo = (bf16_t)(val - (float)hi);
    whi[idx] = hi;
    wlo[idx] = lo;
}

// ---------------------------------------------------------------------------
// IF neuron on raw input x (NCHW f32) -> spikes (NHWC bf16, exact {0,1})
// ---------------------------------------------------------------------------
__global__ void if_neuron(const float* __restrict__ in, bf16_t* __restrict__ spikes) {
    long idx = (long)blockIdx.x * 256 + threadIdx.x;   // ((b*1024+hw)*128+c)
    int  c   = (int)(idx & 127);
    long bh  = idx >> 7;
    int  hw  = (int)(bh & 1023);
    long b   = bh >> 10;
    const long strideT = 32L * 131072L;                // B * C*H*W
    long base  = b * 131072L + (long)c * 1024 + hw;    // NCHW read
    long obase = b * 131072L + (long)hw * 128 + c;     // NHWC write
    float v = 0.0f;
    #pragma unroll
    for (int t = 0; t < 8; ++t) {
        v += in[base + (long)t * strideT];
        float s = (v >= 1.0f) ? 1.0f : 0.0f;
        v = (v >= 1.0f) ? 0.0f : v;
        spikes[obase + (long)t * strideT] = (bf16_t)s;
    }
}

// ---------------------------------------------------------------------------
// Implicit-GEMM 3x3 conv (pad 1) + BN, t-grouped tiling so every lane holds all
// 8 time-steps of one output element -> IF neuron fused in registers (mode 0).
//   Block: b fixed, 16 pixels; M = 8t x 16px = 128 rows, N = 128 cout.
//   8 waves x 16 cout each; acc[t] = 16x16 tile; 16 WMMA/wave/K-step (hi+lo).
//   mode 0: BN + IF -> bf16 spikes NHWC.   mode 1: BN -> f32 NCHW (final).
// ---------------------------------------------------------------------------
__launch_bounds__(256)
__global__ void conv_bn_if(const bf16_t* __restrict__ sp_in,   // NHWC [256][1024][128]
                           const bf16_t* __restrict__ whi,     // [tap][co][ci]
                           const bf16_t* __restrict__ wlo,
                           const float* __restrict__ gamma, const float* __restrict__ beta,
                           const float* __restrict__ mean,  const float* __restrict__ var,
                           void* __restrict__ out_p, int mode) {
    __shared__ __align__(16) unsigned char smem[3 * 2 * 128 * 80];  // 80B padded rows
    unsigned char* As = smem;                 // [2][128][80]  A: rows m = t*16+pl
    unsigned char* Bh = smem + 20480;         // [2][128][80]  W-hi: rows = co
    unsigned char* Bl = smem + 40960;         // [2][128][80]  W-lo

    const int tid = threadIdx.x;
    const int b   = blockIdx.x >> 6;          // batch 0..31
    const int p0  = (blockIdx.x & 63) << 4;   // first of 16 pixels

    const int sm = tid >> 1;                  // staging row 0..127
    const int sh = tid & 1;                   // 16-element half

    const int lane = tid & 31;
    const int wid  = tid >> 5;
    const int nw   = wid * 16;                // wave cout offset
    const int lrow = lane & 15;
    const int hi16 = lane >> 4;

    v8f acc[8];
    #pragma unroll
    for (int t = 0; t < 8; ++t)
        acc[t] = (v8f){0.f, 0.f, 0.f, 0.f, 0.f, 0.f, 0.f, 0.f};

    auto stage = [&](int s, int buf) {
        int tap = s >> 2;                     // 0..8
        int c0  = (s & 3) << 5;               // ci chunk base
        int dy  = tap / 3 - 1, dx = tap % 3 - 1;
        // A: row m = t*16+pl -> spikes[n=t*32+b][yy*32+xx][c0..c0+31], zero pad
        {
            int t  = sm >> 4, pl = sm & 15;
            int p  = p0 + pl;
            int y  = p >> 5, x = p & 31;
            int yy = y + dy, xx = x + dx;
            uint4 d0 = {0u, 0u, 0u, 0u}, d1 = {0u, 0u, 0u, 0u};
            if ((unsigned)yy < 32u && (unsigned)xx < 32u) {
                const uint4* src = (const uint4*)(sp_in +
                    ((long)(t * 32 + b) * 1024 + yy * 32 + xx) * 128 + c0 + sh * 16);
                d0 = src[0]; d1 = src[1];
            }
            uint4* dst = (uint4*)(As + buf * 10240 + sm * 80 + sh * 32);
            dst[0] = d0; dst[1] = d1;
        }
        // B: contiguous copy -> async-to-LDS (no VGPR round trip, ASYNCcnt)
        {
            long widx = (long)(tap * 128 + sm) * 128 + c0 + sh * 16;
            unsigned char* dh = Bh + buf * 10240 + sm * 80 + sh * 32;
            unsigned char* dl = Bl + buf * 10240 + sm * 80 + sh * 32;
            async_copy16(dh,      whi + widx);
            async_copy16(dh + 16, whi + widx + 8);
            async_copy16(dl,      wlo + widx);
            async_copy16(dl + 16, wlo + widx + 8);
        }
    };

    union AB { v16bf v; uint4 q[2]; };

    auto compute = [&](int buf) {
        AB a[8];
        #pragma unroll
        for (int t = 0; t < 8; ++t) {          // A frag for time-step t
            const unsigned char* row = As + buf * 10240 + (t * 16 + lrow) * 80 + hi16 * 32;
            a[t].q[0] = *(const uint4*)(row);
            a[t].q[1] = *(const uint4*)(row + 16);
        }
        AB bhf, blf;
        {
            const unsigned char* rh = Bh + buf * 10240 + (nw + lrow) * 80 + hi16 * 32;
            bhf.q[0] = *(const uint4*)(rh);
            bhf.q[1] = *(const uint4*)(rh + 16);
            const unsigned char* rl = Bl + buf * 10240 + (nw + lrow) * 80 + hi16 * 32;
            blf.q[0] = *(const uint4*)(rl);
            blf.q[1] = *(const uint4*)(rl + 16);
        }
        #pragma unroll
        for (int t = 0; t < 8; ++t) {
            acc[t] = __builtin_amdgcn_wmma_f32_16x16x32_bf16(
                false, a[t].v, false, bhf.v, (short)0, acc[t], false, false);
            acc[t] = __builtin_amdgcn_wmma_f32_16x16x32_bf16(
                false, a[t].v, false, blf.v, (short)0, acc[t], false, false);
        }
    };

    stage(0, 0);
    wait_async0();
    __syncthreads();
    for (int s = 0; s < 36; ++s) {
        if (s + 1 < 36) stage(s + 1, (s + 1) & 1);
        compute(s & 1);
        wait_async0();
        __syncthreads();
    }

    // Epilogue: BN per lane's cout; lane holds all 8 t of (pixel, co) -> fuse IF
    const int co  = nw + lrow;
    const float inv = gamma[co] * rsqrtf(var[co] + EPS_BN);
    const float add = beta[co] - mean[co] * inv;

    if (mode == 0) {
        bf16_t* osp = (bf16_t*)out_p;                    // spikes2, NHWC bf16
        #pragma unroll
        for (int j = 0; j < 8; ++j) {
            int p = p0 + j + hi16 * 8;                   // D: VGPR j -> M=j(+8)
            float v = 0.0f;
            #pragma unroll
            for (int t = 0; t < 8; ++t) {                // sequential IF over t
                v += acc[t][j] * inv + add;
                float sv = (v >= 1.0f) ? 1.0f : 0.0f;
                osp[((long)(t * 32 + b) * 1024 + p) * 128 + co] = (bf16_t)sv;
                v = (v >= 1.0f) ? 0.0f : v;
            }
        }
    } else {
        float* out = (float*)out_p;                      // final, NCHW f32
        #pragma unroll
        for (int j = 0; j < 8; ++j) {
            int p = p0 + j + hi16 * 8;
            #pragma unroll
            for (int t = 0; t < 8; ++t)
                out[((long)(t * 32 + b) * 128 + co) * 1024 + p] = acc[t][j] * inv + add;
        }
    }
}

// ---------------------------------------------------------------------------
extern "C" void kernel_launch(void* const* d_in, const int* in_sizes, int n_in,
                              void* d_out, int out_size, void* d_ws, size_t ws_size,
                              hipStream_t stream) {
    const float* x  = (const float*)d_in[0];
    const float* w1 = (const float*)d_in[1];
    const float* g1 = (const float*)d_in[2];
    const float* b1 = (const float*)d_in[3];
    const float* m1 = (const float*)d_in[4];
    const float* v1 = (const float*)d_in[5];
    const float* w2 = (const float*)d_in[6];
    const float* g2 = (const float*)d_in[7];
    const float* b2 = (const float*)d_in[8];
    const float* m2 = (const float*)d_in[9];
    const float* v2 = (const float*)d_in[10];

    unsigned char* ws = (unsigned char*)d_ws;
    bf16_t* spikes1 = (bf16_t*)(ws);                       // 67,108,864 B
    bf16_t* spikes2 = (bf16_t*)(ws + 67108864);            // 67,108,864 B
    bf16_t* wh1 = (bf16_t*)(ws + 134217728);               // 294,912 B each
    bf16_t* wl1 = (bf16_t*)(ws + 134217728 + 294912);
    bf16_t* wh2 = (bf16_t*)(ws + 134217728 + 2 * 294912);
    bf16_t* wl2 = (bf16_t*)(ws + 134217728 + 3 * 294912);

    prep_weights<<<576, 256, 0, stream>>>(w1, wh1, wl1);
    prep_weights<<<576, 256, 0, stream>>>(w2, wh2, wl2);

    // IF1: x (NCHW f32) -> spikes1 (NHWC bf16)
    if_neuron<<<16384, 256, 0, stream>>>(x, spikes1);
    // conv1 + BN1 + fused IF2 -> spikes2 (NHWC bf16); no f32 intermediate
    conv_bn_if<<<2048, 256, 0, stream>>>(spikes1, wh1, wl1, g1, b1, m1, v1,
                                         (void*)spikes2, 0);
    // conv2 + BN2 -> d_out (NCHW f32, reference layout)
    conv_bn_if<<<2048, 256, 0, stream>>>(spikes2, wh2, wl2, g2, b2, m2, v2,
                                         d_out, 1);
}
